// VectorQuantizer_23424751632716
// MI455X (gfx1250) — matrix operand order repaired and checked by reading, not compiled, not run on hardware
//
#include <hip/hip_runtime.h>

typedef __attribute__((ext_vector_type(16))) _Float16 v16h;
typedef __attribute__((ext_vector_type(8)))  float    v8f;

#define N_TOK 262144
#define DIM   64
#define K_CB  1024
#define TOK_PER_WAVE   16
#define WAVES_PER_BLK  8
#define TOK_PER_BLK    (TOK_PER_WAVE * WAVES_PER_BLK)

// Dynamic LDS layout:
//   [0, 131072)        : cbn f16 in B-fragment order: half index = ((t*2+ks)*32 + lane)*16 + j
//   [131072, 163840)   : per-wave xn fp32 tiles, wave w at +w*16*64 floats

#define WMMA_F16(A, B, C) \
    __builtin_amdgcn_wmma_f32_16x16x32_f16(false, (A), false, (B), (short)0, (C), false, false)

extern "C" __global__ __launch_bounds__(256)
void vq_wmma_kernel(const float* __restrict__ x, const float* __restrict__ cbraw,
                    float* __restrict__ out_zq, float* __restrict__ out_z,
                    float* __restrict__ out_xn, float* __restrict__ out_idx)
{
    extern __shared__ char smem[];
    _Float16* cbnFrag = (_Float16*)smem;                 // 65536 halves = 128KB
    float*    xnBase  = (float*)(smem + 131072);         // 8192 floats  = 32KB

    const int tid  = threadIdx.x;
    const int lane = tid & 31;
    const int wave = tid >> 5;
    const int h    = lane >> 4;    // half-group within wave
    const int nl   = lane & 15;

    // ---------------- Stage codebook: double-normalize, scatter to B-fragment layout ----------------
    // cb  = raw / (||raw|| + 1e-8) ; cbn = cb / (||cb|| + 1e-8)  => scale = 1/(n1*n2)
    for (int row = wave; row < K_CB; row += WAVES_PER_BLK) {
        float2 v = ((const float2*)(cbraw + row * DIM))[lane];   // d = 2*lane, 2*lane+1
        float ss = v.x * v.x + v.y * v.y;
        #pragma unroll
        for (int off = 16; off > 0; off >>= 1) ss += __shfl_xor(ss, off, 32);
        float rt = sqrtf(ss);
        float n1 = rt + 1e-8f;
        float n2 = rt / n1 + 1e-8f;
        float scale = 1.0f / (n1 * n2);

        int d0 = lane * 2;
        int t  = row >> 4, n = row & 15;
        int ks = d0 >> 5;
        int dd = d0 & 31;
        int bh = dd >> 4, j = dd & 15;              // j is even; j and j+1 in same block
        int blane = bh * 16 + n;
        int addr  = ((t * 2 + ks) * 32 + blane) * 16 + j;   // half units, dword aligned
        union { _Float16 hh[2]; unsigned int u; } pk;
        pk.hh[0] = (_Float16)(v.x * scale);
        pk.hh[1] = (_Float16)(v.y * scale);
        *(unsigned int*)(cbnFrag + addr) = pk.u;
    }
    __syncthreads();

    // ---------------- Normalize this wave's 16 token rows (fp32) ----------------
    const int tokBase = blockIdx.x * TOK_PER_BLK + wave * TOK_PER_WAVE;
    float* xnTile = xnBase + wave * (TOK_PER_WAVE * DIM);

    if (lane < 16) {
        const float* xrow = x + (size_t)(tokBase + lane) * DIM;
        float ss = 0.f;
        #pragma unroll
        for (int i = 0; i < 16; ++i) {
            float4 q = ((const float4*)xrow)[i];
            ss += q.x * q.x + q.y * q.y + q.z * q.z + q.w * q.w;
        }
        float inv = 1.0f / (sqrtf(ss) + 1e-8f);
        float* oxn = out_xn + (size_t)(tokBase + lane) * DIM;
        #pragma unroll
        for (int i = 0; i < 16; ++i) {
            float4 q = ((const float4*)xrow)[i];
            q.x *= inv; q.y *= inv; q.z *= inv; q.w *= inv;
            ((float4*)(xnTile + lane * DIM))[i] = q;
            ((float4*)oxn)[i] = q;
        }
    }

    // ---------------- Build A fragments (16x32, contiguous k = 16h + s, same order as B) ----------------
    // lane l: m = l&15; K = {kb..kb+15} with kb = (l>>4)*16
    v16h A0, A1;
    {
        const float* xr = xnTile + nl * DIM;
        const int kb = h * 16;
        #pragma unroll
        for (int j = 0; j < 8; ++j) {
            A0[j]     = (_Float16)xr[kb + j];
            A0[j + 8] = (_Float16)xr[kb + 8 + j];
            A1[j]     = (_Float16)xr[kb + 32 + j];
            A1[j + 8] = (_Float16)xr[kb + 40 + j];
        }
    }

    // ---------------- K loop: 64 column tiles, 2 WMMAs each ----------------
    // Software-pipelined argmax: compare tile t-1 results while tile t's WMMAs run,
    // so WMMA->VALU consumers are >1 tile away (no hazard NOPs, full co-execution).
    float bestVal[8];
    int   bestIdx[8];
    #pragma unroll
    for (int r = 0; r < 8; ++r) { bestVal[r] = -3.4e38f; bestIdx[r] = 0; }

    const _Float16* fbase = cbnFrag + lane * 16;
    v8f accP;
    {   // prologue: tile 0
        v16h B0 = *(const v16h*)(fbase);
        v16h B1 = *(const v16h*)(fbase + 32 * 16);
        v8f acc = {};
        acc  = WMMA_F16(A0, B0, acc);
        accP = WMMA_F16(A1, B1, acc);
    }

    #pragma unroll 2
    for (int t = 1; t < K_CB / 16; ++t) {
        const _Float16* fp = fbase + t * (2 * 32 * 16);
        v16h B0 = *(const v16h*)(fp);
        v16h B1 = *(const v16h*)(fp + 32 * 16);
        v8f acc = {};
        acc = WMMA_F16(A0, B0, acc);
        acc = WMMA_F16(A1, B1, acc);
        // consume previous tile's scores (independent of in-flight WMMAs)
        const int col = (t - 1) * 16 + nl;
        #pragma unroll
        for (int r = 0; r < 8; ++r) {
            if (accP[r] > bestVal[r]) { bestVal[r] = accP[r]; bestIdx[r] = col; }
        }
        accP = acc;
    }
    {   // drain: last tile's scores
        const int col = (K_CB / 16 - 1) * 16 + nl;
        #pragma unroll
        for (int r = 0; r < 8; ++r) {
            if (accP[r] > bestVal[r]) { bestVal[r] = accP[r]; bestIdx[r] = col; }
        }
    }

    // ---------------- Cross-lane argmax within each 16-lane half (rows r + 8h) ----------------
    #pragma unroll
    for (int r = 0; r < 8; ++r) {
        #pragma unroll
        for (int off = 8; off > 0; off >>= 1) {
            float ov = __shfl_xor(bestVal[r], off, 32);
            int   oi = __shfl_xor(bestIdx[r], off, 32);
            if (ov > bestVal[r] || (ov == bestVal[r] && oi < bestIdx[r])) {
                bestVal[r] = ov; bestIdx[r] = oi;
            }
        }
    }

    // ---------------- Indices (as float, matching concatenated f32 output) ----------------
    if (nl < 8) {
        out_idx[tokBase + h * 8 + nl] = (float)bestIdx[nl];
    }

    // ---------------- Gather z = normalize(codebook)[idx]; z_q = xn + (z - xn) ----------------
    #pragma unroll
    for (int r = 0; r < 8; ++r) {
        const int row = h * 8 + r;
        const int idx = bestIdx[r];                       // uniform within the 16-lane group
        float4 raw = ((const float4*)(cbraw + idx * DIM))[nl];
        float ss = raw.x * raw.x + raw.y * raw.y + raw.z * raw.z + raw.w * raw.w;
        #pragma unroll
        for (int off = 8; off > 0; off >>= 1) ss += __shfl_xor(ss, off, 32);
        float inv = 1.0f / (sqrtf(ss) + 1e-8f);
        float4 zv;
        zv.x = raw.x * inv; zv.y = raw.y * inv; zv.z = raw.z * inv; zv.w = raw.w * inv;
        float4 xv = ((const float4*)(xnTile + row * DIM))[nl];
        float4 zq;  // xn + (z - xn), reference op order
        zq.x = xv.x + (zv.x - xv.x);
        zq.y = xv.y + (zv.y - xv.y);
        zq.z = xv.z + (zv.z - xv.z);
        zq.w = xv.w + (zv.w - xv.w);
        size_t o = (size_t)(tokBase + row) * DIM + nl * 4;
        *(float4*)(out_z  + o) = zv;
        *(float4*)(out_zq + o) = zq;
    }
}

extern "C" void kernel_launch(void* const* d_in, const int* in_sizes, int n_in,
                              void* d_out, int out_size, void* d_ws, size_t ws_size,
                              hipStream_t stream) {
    (void)in_sizes; (void)n_in; (void)out_size; (void)d_ws; (void)ws_size;
    const float* x  = (const float*)d_in[0];
    const float* cb = (const float*)d_in[1];
    // d_in[2] = training flag (0) -> identity dropout, ignored.
    float* out = (float*)d_out;
    const size_t ND = (size_t)N_TOK * DIM;
    float* out_zq  = out;
    float* out_z   = out + ND;
    float* out_xn  = out + 2 * ND;
    float* out_idx = out + 3 * ND;

    dim3 grid(N_TOK / TOK_PER_BLK);   // 2048 blocks
    dim3 block(256);                  // 8 waves (wave32)
    size_t lds = 131072 + 32768;      // 160KB dynamic LDS -> 2 blocks per WGP
    vq_wmma_kernel<<<grid, block, lds, stream>>>(x, cb, out_zq, out_z, out_xn, out_idx);
}
